// AttnDecoderRNN_41059887350368
// MI455X (gfx1250) — compile-verified, matrix operand order checked by reading, not run on hardware
//
#include <hip/hip_runtime.h>
#include <hip/hip_bf16.h>

// ---------------------------------------------------------------------------
// AttnDecoderRNN for MI455X (gfx1250, wave32, WMMA).
// B=128, T=512, H=512, D=128, OUT_LEN=256.
// bf16 storage + f32 accumulate: keeps the two 33.5M-element streamed tensors
// (enc, UaK) at 64MB each -> both resident in the 192MB L2, and feeds
// v_wmma_f32_16x16x32_bf16 for all GEMMs.
// GEMM uses 64x32 register blocking per wave (8 WMMA per 6 fragment loads).
// ---------------------------------------------------------------------------

typedef __attribute__((ext_vector_type(16))) __bf16 v16bf;
typedef __attribute__((ext_vector_type(8)))  float  v8f;

#define BB 128
#define TT 512
#define HH 512
#define DD 128
#define OUTLEN 256
#define XK 1536            // [latent(512) | ctx(512) | h(512)]

__device__ __forceinline__ float sigmoidf_(float x) {
    return 1.0f / (1.0f + __expf(-x));
}

// ------------------------- prep kernels ------------------------------------

__global__ void cvt_f32_to_bf16(const float* __restrict__ src,
                                __bf16* __restrict__ dst, int n) {
    int i = blockIdx.x * 256 + threadIdx.x;
    if (i < n) dst[i] = (__bf16)src[i];
}

// dst[h*HH + g] = src[g*HH + h]   (Wa -> WaT, so phase-1 reads coalesce)
__global__ void transpose_cvt_bf16(const float* __restrict__ src,
                                   __bf16* __restrict__ dst, int n) {
    int i = blockIdx.x * 256 + threadIdx.x;
    if (i >= n) return;
    int h = i / HH, g = i % HH;
    dst[i] = (__bf16)src[g * HH + h];
}

// Wcat (4H x 1536) = [W_ih (4H x 1024) | W_hh (4H x 512)] in bf16
__global__ void build_wcat(const float* __restrict__ wih,
                           const float* __restrict__ whh,
                           __bf16* __restrict__ dst, int n) {
    int i = blockIdx.x * 256 + threadIdx.x;
    if (i >= n) return;
    int r = i / XK, k = i % XK;
    float v = (k < 2 * HH) ? wih[r * (2 * HH) + k] : whh[r * HH + (k - 2 * HH)];
    dst[i] = (__bf16)v;
}

// X[b][0:512] = latent_h (constant each step), rest zeroed (ctx, h0=0)
__global__ void init_x(const float* __restrict__ latent,
                       __bf16* __restrict__ X, int n) {
    int i = blockIdx.x * 256 + threadIdx.x;
    if (i >= n) return;
    int b = i / XK, j = i % XK;
    X[i] = (j < HH) ? (__bf16)latent[b * HH + j] : (__bf16)0.0f;
}

__global__ void zero_f32(float* __restrict__ p, int n) {
    int i = blockIdx.x * 256 + threadIdx.x;
    if (i < n) p[i] = 0.0f;
}

// ------------------------- WMMA GEMM, 64x32 per wave -----------------------
// C[m][n] = sum_k A[m][k] * W[n][k] (+ bias0[n] + bias1[n])
// A: M x K bf16 row-major (lda), W: N x K bf16 row-major (ldw).
// Each wave owns a 64x32 macro-tile: 4 M-tiles x 2 N-tiles, K consumed 32 at
// a time. Per K-step: 4 A-frags + 2 B-frags -> 8 v_wmma_f32_16x16x32_bf16.
// Requires M % 64 == 0, N % 32 == 0, K % 32 == 0.
// Output fp32 (Cf) or bf16 (Cb).

__global__ void wmma_gemm_bf16(const __bf16* __restrict__ A, int lda,
                               const __bf16* __restrict__ W, int ldw,
                               float* __restrict__ Cf, __bf16* __restrict__ Cb,
                               long long ldc, long long cbase,
                               const float* __restrict__ bias0,
                               const float* __restrict__ bias1,
                               int M, int N, int K) {
    const int lane = threadIdx.x & 31;
    const int wid  = blockIdx.x * (blockDim.x >> 5) + (threadIdx.x >> 5);
    const int grpN = N >> 5;                 // N-groups of 32
    const int gm = wid / grpN;
    const int gn = wid % grpN;
    const int m0 = gm << 6;                  // 64 rows
    const int n0 = gn << 5;                  // 32 cols
    if (m0 >= M) return;

    const int rlane = lane & 15;             // row within fragment
    const int khalf = (lane >> 4) << 4;      // lanes 16-31 carry K+16 half

    const __bf16* arow = A + (size_t)(m0 + rlane) * lda + khalf;
    const __bf16* wrow = W + (size_t)(n0 + rlane) * ldw + khalf;
    const size_t astep = (size_t)16 * lda;   // next M tile
    const size_t wstep = (size_t)16 * ldw;   // next N tile

    v8f acc[4][2] = {};
    for (int k0 = 0; k0 < K; k0 += 32) {
        v16bf a[4], b[2];
#pragma unroll
        for (int mi = 0; mi < 4; ++mi)
            a[mi] = *(const v16bf*)(arow + (size_t)mi * astep + k0);
#pragma unroll
        for (int ni = 0; ni < 2; ++ni)
            b[ni] = *(const v16bf*)(wrow + (size_t)ni * wstep + k0);
        if (k0 + 32 < K) {                   // global_prefetch_b8
            __builtin_prefetch(arow + k0 + 32, 0, 1);
            __builtin_prefetch(wrow + k0 + 32, 0, 1);
        }
#pragma unroll
        for (int mi = 0; mi < 4; ++mi)
#pragma unroll
            for (int ni = 0; ni < 2; ++ni)
                acc[mi][ni] = __builtin_amdgcn_wmma_f32_16x16x32_bf16(
                    false, a[mi], false, b[ni], (short)0, acc[mi][ni],
                    false, false);
    }

    // C/D layout: VGPR i -> M = i (lanes 0-15) / i+8 (lanes 16-31), N = lane%16
    const int crow = (lane >> 4) << 3;
    const int ccol = lane & 15;
#pragma unroll
    for (int ni = 0; ni < 2; ++ni) {
        const long long col = n0 + ni * 16 + ccol;
        float bias = 0.0f;
        if (bias0) bias += bias0[col];
        if (bias1) bias += bias1[col];
#pragma unroll
        for (int mi = 0; mi < 4; ++mi) {
#pragma unroll
            for (int i = 0; i < 8; ++i) {
                long long m = m0 + mi * 16 + crow + i;
                float v = acc[mi][ni][i] + bias;
                if (Cb) Cb[cbase + m * ldc + col] = (__bf16)v;
                else    Cf[cbase + m * ldc + col] = v;
            }
        }
    }
}

// ------------------- fused attention: q, tanh-score, softmax ---------------
// one block (512 threads) per batch row b
__global__ void attn_kernel(const __bf16* __restrict__ X,     // h in block 2
                            const __bf16* __restrict__ WaT,   // [h][g]
                            const float* __restrict__ Va,     // (H,)
                            const __bf16* __restrict__ UaK,   // (B*T, H) bf16
                            float* __restrict__ wout) {       // (B, T)
    __shared__ float hsL[HH], vaL[HH], qL[HH], rL[TT];
    const int b = blockIdx.x;
    const int t = threadIdx.x;

    hsL[t] = (float)X[(size_t)b * XK + 2 * HH + t];
    vaL[t] = Va[t];
    __syncthreads();

    // phase 1: q[g=t] = sum_h hs[h] * WaT[h][g]   (coalesced in g)
    float q = 0.0f;
#pragma unroll 4
    for (int h = 0; h < HH; ++h) q += hsL[h] * (float)WaT[h * HH + t];
    qL[t] = q;
    __syncthreads();

    // phase 2: score[t] = sum_h va[h] * tanh(q[h] + UaK[b,t,h])
    float s = 0.0f;
    const v16bf* up = (const v16bf*)(UaK + ((size_t)b * TT + t) * HH);
    for (int h0 = 0; h0 < HH; h0 += 16) {
        v16bf u = up[h0 >> 4];
#pragma unroll
        for (int j = 0; j < 16; ++j)
            s += vaL[h0 + j] * tanhf(qL[h0 + j] + (float)u[j]);
    }

    // phase 3: softmax over T within the block
    rL[t] = s;
    __syncthreads();
    for (int st = TT >> 1; st > 0; st >>= 1) {
        if (t < st) rL[t] = fmaxf(rL[t], rL[t + st]);
        __syncthreads();
    }
    float mx = rL[0];
    __syncthreads();
    float e = __expf(s - mx);
    rL[t] = e;
    __syncthreads();
    for (int st = TT >> 1; st > 0; st >>= 1) {
        if (t < st) rL[t] += rL[t + st];
        __syncthreads();
    }
    wout[(size_t)b * TT + t] = e / rL[0];
}

// ------------------- context: ctx[b,h] = sum_t w[b,t]*enc[b,t,h] -----------
// grid (H/256, B), writes bf16 straight into X's ctx block
__global__ void ctx_kernel(const float* __restrict__ wattn,
                           const __bf16* __restrict__ enc,
                           __bf16* __restrict__ X) {
    __shared__ float wL[TT];
    const int b = blockIdx.y;
    const int h = blockIdx.x * 256 + threadIdx.x;
    wL[threadIdx.x]       = wattn[(size_t)b * TT + threadIdx.x];
    wL[threadIdx.x + 256] = wattn[(size_t)b * TT + threadIdx.x + 256];
    __syncthreads();
    float acc = 0.0f;
    const __bf16* ep = enc + (size_t)b * TT * HH + h;
#pragma unroll 4
    for (int t = 0; t < TT; ++t) acc += wL[t] * (float)ep[(size_t)t * HH];
    X[(size_t)b * XK + HH + h] = (__bf16)acc;
}

// ------------------- LSTM cell elementwise ---------------------------------
__global__ void cell_kernel(const float* __restrict__ gates,  // (B, 4H)
                            float* __restrict__ c,            // (B, H)
                            __bf16* __restrict__ X) {         // h -> block 2
    int idx = blockIdx.x * 256 + threadIdx.x;      // 0 .. B*H-1
    int b = idx >> 9, h = idx & (HH - 1);
    const float* g = gates + (size_t)b * 4 * HH;
    float gi = sigmoidf_(g[h]);
    float gf = sigmoidf_(g[HH + h]);
    float gg = tanhf(g[2 * HH + h]);
    float go = sigmoidf_(g[3 * HH + h]);
    float cn = gf * c[idx] + gi * gg;
    float hn = go * tanhf(cn);
    c[idx] = cn;
    X[(size_t)b * XK + 2 * HH + h] = (__bf16)hn;
}

// ---------------------------------------------------------------------------

extern "C" void kernel_launch(void* const* d_in, const int* in_sizes, int n_in,
                              void* d_out, int out_size, void* d_ws, size_t ws_size,
                              hipStream_t stream) {
    const float* enc_f   = (const float*)d_in[0];   // (B,T,H)
    const float* latent  = (const float*)d_in[1];   // (B,H)
    const float* Wa      = (const float*)d_in[2];   // (H,H)
    const float* Ua      = (const float*)d_in[3];   // (H,H)
    const float* Va      = (const float*)d_in[4];   // (1,H)
    const float* W_ih    = (const float*)d_in[5];   // (4H,2H)
    const float* W_hh    = (const float*)d_in[6];   // (4H,H)
    const float* b_ih    = (const float*)d_in[7];   // (4H,)
    const float* b_hh    = (const float*)d_in[8];   // (4H,)
    const float* Wp      = (const float*)d_in[9];   // (D,H)
    const float* bp      = (const float*)d_in[10];  // (D,)
    float* out           = (float*)d_out;           // (B, OUT_LEN, D)

    // ---- workspace carve-up (256B aligned) ----
    size_t off = 0;
    auto carve = [&](size_t bytes) -> void* {
        off = (off + 255) & ~(size_t)255;
        void* p = (char*)d_ws + off;
        off += bytes;
        return p;
    };
    __bf16* enc_b  = (__bf16*)carve((size_t)BB * TT * HH * 2);   // 64 MB
    __bf16* UaKb   = (__bf16*)carve((size_t)BB * TT * HH * 2);   // 64 MB
    __bf16* Ua_b   = (__bf16*)carve((size_t)HH * HH * 2);
    __bf16* WaT    = (__bf16*)carve((size_t)HH * HH * 2);
    __bf16* Wcat   = (__bf16*)carve((size_t)4 * HH * XK * 2);    // 6 MB
    __bf16* Wp_b   = (__bf16*)carve((size_t)DD * HH * 2);
    __bf16* X      = (__bf16*)carve((size_t)BB * XK * 2);
    float*  wattn  = (float*)carve((size_t)BB * TT * 4);
    float*  gates  = (float*)carve((size_t)BB * 4 * HH * 4);
    float*  cst    = (float*)carve((size_t)BB * HH * 4);
    (void)ws_size; (void)in_sizes; (void)n_in; (void)out_size;

    // ---- one-time prep ----
    {
        int n = BB * TT * HH;
        cvt_f32_to_bf16<<<(n + 255) / 256, 256, 0, stream>>>(enc_f, enc_b, n);
    }
    cvt_f32_to_bf16<<<(HH * HH + 255) / 256, 256, 0, stream>>>(Ua, Ua_b, HH * HH);
    transpose_cvt_bf16<<<(HH * HH + 255) / 256, 256, 0, stream>>>(Wa, WaT, HH * HH);
    {
        int n = 4 * HH * XK;
        build_wcat<<<(n + 255) / 256, 256, 0, stream>>>(W_ih, W_hh, Wcat, n);
    }
    cvt_f32_to_bf16<<<(DD * HH + 255) / 256, 256, 0, stream>>>(Wp, Wp_b, DD * HH);
    {
        int n = BB * XK;
        init_x<<<(n + 255) / 256, 256, 0, stream>>>(latent, X, n);
    }
    zero_f32<<<(BB * HH + 255) / 256, 256, 0, stream>>>(cst, BB * HH);

    // waves = (M/64)*(N/32); 8 waves per 256-thread block
    auto gemm_blocks = [](int M, int N) -> int {
        int waves = (M >> 6) * (N >> 5);
        return (waves + 7) / 8;
    };

    // ---- hoisted UaK = enc @ Ua^T : M=B*T, N=H, K=H, bf16 out ----
    {
        int M = BB * TT, N = HH, K = HH;
        wmma_gemm_bf16<<<gemm_blocks(M, N), 256, 0, stream>>>(
            enc_b, HH, Ua_b, HH,
            nullptr, UaKb, (long long)HH, 0LL,
            nullptr, nullptr, M, N, K);
    }

    // ---- 256 decode steps ----
    for (int s = 0; s < OUTLEN; ++s) {
        // attention: q + tanh-score + softmax  (one WG per batch row)
        attn_kernel<<<BB, TT, 0, stream>>>(X, WaT, Va, UaKb, wattn);

        // context -> X ctx block (bf16)
        ctx_kernel<<<dim3(HH / 256, BB), 256, 0, stream>>>(wattn, enc_b, X);

        // gates = X @ Wcat^T + b_ih + b_hh : M=128, N=2048, K=1536
        {
            int M = BB, N = 4 * HH, K = XK;
            wmma_gemm_bf16<<<gemm_blocks(M, N), 256, 0, stream>>>(
                X, XK, Wcat, XK,
                gates, nullptr, (long long)(4 * HH), 0LL,
                b_ih, b_hh, M, N, K);
        }

        // LSTM cell; writes h (bf16) back into X block 2
        cell_kernel<<<(BB * HH) / 256, 256, 0, stream>>>(gates, cst, X);

        // y = h @ Wp^T + bp : M=128, N=128, K=512 -> out[b, s, :]
        {
            int M = BB, N = DD, K = HH;
            wmma_gemm_bf16<<<gemm_blocks(M, N), 256, 0, stream>>>(
                X + 2 * HH, XK, Wp_b, HH,
                out, nullptr, (long long)(OUTLEN * DD), (long long)s * DD,
                bp, nullptr, M, N, K);
        }
    }
}
